// BitLinear_33792802685052
// MI455X (gfx1250) — 8-bit matrix operand order repaired and checked by reading, compiled, not run on hardware
//
#include <hip/hip_runtime.h>
#include <hip/hip_bf16.h>
#include <cstdint>
#include <cstddef>


typedef __attribute__((ext_vector_type(8))) int   v8i;
typedef __attribute__((ext_vector_type(4))) float v4f;

#define K_DIM 2048
#define N_DIM 2048
#define M_TOK 16384

__global__ void __launch_bounds__(256) quant_act(const float* __restrict__ x,
                                                 int8_t* __restrict__ xq,
                                                 float* __restrict__ dq) {
    __shared__ float red[256];
    const int row = blockIdx.x;
    const int t   = threadIdx.x;

    const v4f* xr = (const v4f*)(x + (size_t)row * K_DIM + t * 8);
    const v4f v0 = __builtin_nontemporal_load(xr);
    const v4f v1 = __builtin_nontemporal_load(xr + 1);

    float m = fmaxf(fmaxf(fmaxf(fabsf(v0.x), fabsf(v0.y)),
                          fmaxf(fabsf(v0.z), fabsf(v0.w))),
                    fmaxf(fmaxf(fabsf(v1.x), fabsf(v1.y)),
                          fmaxf(fabsf(v1.z), fabsf(v1.w))));
    red[t] = m;
    __syncthreads();
    #pragma unroll
    for (int s = 128; s > 0; s >>= 1) {
        if (t < s) red[t] = fmaxf(red[t], red[t + s]);
        __syncthreads();
    }

    const float amax  = fmaxf(red[0], 1e-5f);
    const float scale = 127.0f / amax;

    float f[8] = {v0.x, v0.y, v0.z, v0.w, v1.x, v1.y, v1.z, v1.w};
    int q[8];
    #pragma unroll
    for (int i = 0; i < 8; ++i) {
        float qf = rintf(f[i] * scale);
        qf = fminf(fmaxf(qf, -128.0f), 127.0f);
        q[i] = (int)qf;
    }
    const int p0 = (q[0] & 255) | ((q[1] & 255) << 8) |
                   ((q[2] & 255) << 16) | (q[3] << 24);
    const int p1 = (q[4] & 255) | ((q[5] & 255) << 8) |
                   ((q[6] & 255) << 16) | (q[7] << 24);
    int* xqi = (int*)(xq + (size_t)row * K_DIM) + t * 2;
    xqi[0] = p0;
    xqi[1] = p1;

    if (t == 0) dq[row] = amax * (1.0f / 127.0f);
}

__global__ void __launch_bounds__(256) unpack_w(const int* __restrict__ wp,
                                                int8_t* __restrict__ w8) {
    const int idx = blockIdx.x * 256 + threadIdx.x;
    const int r = idx >> 11;
    const int c = idx & 2047;
    const int p = __builtin_nontemporal_load(wp + idx);
    #pragma unroll
    for (int i = 0; i < 4; ++i) {
        w8[(size_t)((i << 9) + r) * K_DIM + c] =
            (int8_t)(((p >> (2 * i)) & 3) - 1);
    }
}

__global__ void __launch_bounds__(256) bitnet_gemm(
    const int8_t* __restrict__ xq, const int8_t* __restrict__ w8,
    const float*  __restrict__ dq, const float* __restrict__ wscale,
    const float*  __restrict__ bias, float* __restrict__ out) {

    const int lane = threadIdx.x & 31;
    const int wave = threadIdx.x >> 5;
    const int l15  = lane & 15;
    const int half = lane >> 4;

    const int m0 = blockIdx.y * 128 + (wave >> 1) * 32;
    const int n0 = blockIdx.x * 64  + (wave & 1) * 32;

    int ka[8], kb[8];
    #pragma unroll
    for (int v = 0; v < 8; ++v) {
        ka[v] = ((v & 4) << 3) + (v & 3) * 4 + half * 16;
        kb[v] = ((v & 4) << 3) + (v & 3) * 4 + half * 16;
    }

    const int8_t* a0p = xq + (size_t)(m0 + l15) * K_DIM;
    const int8_t* a1p = a0p + (size_t)16 * K_DIM;
    const int8_t* b0p = w8 + (size_t)(n0 + l15) * K_DIM;
    const int8_t* b1p = b0p + (size_t)16 * K_DIM;

    v8i c00 = {}, c01 = {}, c10 = {}, c11 = {};

    for (int k0 = 0; k0 < K_DIM; k0 += 64) {
        v8i a0, a1, b0, b1;
        #pragma unroll
        for (int v = 0; v < 8; ++v) {
            a0[v] = *(const int*)(a0p + k0 + ka[v]);
            a1[v] = *(const int*)(a1p + k0 + ka[v]);
            b0[v] = *(const int*)(b0p + k0 + kb[v]);
            b1[v] = *(const int*)(b1p + k0 + kb[v]);
        }
        __builtin_prefetch(a0p + k0 + 64, 0, 3);
        __builtin_prefetch(b0p + k0 + 64, 0, 3);

        c00 = __builtin_amdgcn_wmma_i32_16x16x64_iu8(true, a0, true, b0, c00,
                                                     false, false);
        c01 = __builtin_amdgcn_wmma_i32_16x16x64_iu8(true, a0, true, b1, c01,
                                                     false, false);
        c10 = __builtin_amdgcn_wmma_i32_16x16x64_iu8(true, a1, true, b0, c10,
                                                     false, false);
        c11 = __builtin_amdgcn_wmma_i32_16x16x64_iu8(true, a1, true, b1, c11,
                                                     false, false);
    }

    const float inv_ws = 1.0f / wscale[0];
    const float bn0 = bias[n0 + l15];
    const float bn1 = bias[n0 + 16 + l15];

    #pragma unroll
    for (int r = 0; r < 8; ++r) {
        const int row0 = m0 + half * 8 + r;
        const int row1 = row0 + 16;
        const float s0 = dq[row0] * inv_ws;
        const float s1 = dq[row1] * inv_ws;
        __builtin_nontemporal_store((float)c00[r] * s0 + bn0,
                                    out + (size_t)row0 * N_DIM + n0 + l15);
        __builtin_nontemporal_store((float)c01[r] * s0 + bn1,
                                    out + (size_t)row0 * N_DIM + n0 + 16 + l15);
        __builtin_nontemporal_store((float)c10[r] * s1 + bn0,
                                    out + (size_t)row1 * N_DIM + n0 + l15);
        __builtin_nontemporal_store((float)c11[r] * s1 + bn1,
                                    out + (size_t)row1 * N_DIM + n0 + 16 + l15);
    }
}

extern "C" void kernel_launch(void* const* d_in, const int* in_sizes, int n_in,
                              void* d_out, int out_size, void* d_ws,
                              size_t ws_size, hipStream_t stream) {
    const float* x      = (const float*)d_in[0];
    const int*   wp     = (const int*)d_in[1];
    const float* wscale = (const float*)d_in[2];
    const float* bias   = (const float*)d_in[3];
    float*       out    = (float*)d_out;

    int8_t* xq = (int8_t*)d_ws;
    float*  dq = (float*)((char*)d_ws + (size_t)M_TOK * K_DIM);
    int8_t* w8 = (int8_t*)((char*)d_ws + (size_t)M_TOK * K_DIM +
                           (size_t)M_TOK * sizeof(float));

    quant_act<<<M_TOK, 256, 0, stream>>>(x, xq, dq);
    unpack_w<<<(512 * 2048) / 256, 256, 0, stream>>>(wp, w8);

    dim3 grid(N_DIM / 64, M_TOK / 128);
    bitnet_gemm<<<grid, 256, 0, stream>>>(xq, w8, dq, wscale, bias, out);
}
